// Net_63531156242917
// MI455X (gfx1250) — compile-verified, matrix operand order checked by reading, not run on hardware
//
#include <hip/hip_runtime.h>
#include <hip/hip_bf16.h>

// ---------------------------------------------------------------------------
// SNN forward (4-layer LIF), restructured as 4 big WMMA GEMMs + 4 LIF scans.
//   T=50, B=256, D=768, H=2048, O=10.  M = T*B = 12800.
// d_out layout (fp32, flat, in reference return order):
//   s1,s2,s3 [T,B,H]  s4 [T,B,O]  m1..m3 [T,B,H]  m4 [T,B,O]  c1..c3  c4
// Workspace: f16 copies of W1..W3 and zero-padded W4 (N padded 10 -> 16).
// ---------------------------------------------------------------------------

typedef _Float16 f16x8  __attribute__((ext_vector_type(8)));
typedef _Float16 f16x16 __attribute__((ext_vector_type(16)));
typedef float    f32x4  __attribute__((ext_vector_type(4)));
typedef float    f32x8  __attribute__((ext_vector_type(8)));

// ---------------------------------------------------------------------------
// fp32 -> f16 weight conversion (grid-stride not needed; exact grids)
// ---------------------------------------------------------------------------
__global__ void cvt_f32_to_f16(const float* __restrict__ src,
                               _Float16* __restrict__ dst, int n) {
  int i = blockIdx.x * blockDim.x + threadIdx.x;
  if (i < n) dst[i] = (_Float16)src[i];
}

// W4 [O,K] -> padded [NP,K] with zero rows for n >= O
__global__ void cvt_pad_f16(const float* __restrict__ src,
                            _Float16* __restrict__ dst,
                            int K, int Nreal, int Npad) {
  int i = blockIdx.x * blockDim.x + threadIdx.x;
  if (i < Npad * K) {
    int n = i / K;
    dst[i] = (n < Nreal) ? (_Float16)src[i] : (_Float16)0.0f;
  }
}

// ---------------------------------------------------------------------------
// A-fragment loader: fp32 source row, convert to f16 in regs.
// Lane l<16 holds row m=l, K = {k0..k0+7, k0+16..k0+23};
// lane l>=16 holds row m=l-16, K = {k0+8..k0+15, k0+24..k0+31}.
// ---------------------------------------------------------------------------
__device__ __forceinline__ f16x16 load_a_frag(const float* __restrict__ p) {
  f32x4 a0 = *(const f32x4*)(p);
  f32x4 a1 = *(const f32x4*)(p + 4);
  f32x4 a2 = *(const f32x4*)(p + 16);
  f32x4 a3 = *(const f32x4*)(p + 20);
  f16x16 r;
#pragma unroll
  for (int e = 0; e < 4; ++e) {
    r[e]      = (_Float16)a0[e];
    r[4 + e]  = (_Float16)a1[e];
    r[8 + e]  = (_Float16)a2[e];
    r[12 + e] = (_Float16)a3[e];
  }
  return r;
}

// B-fragment loader: f16 weight row (symmetric K striping -> computes A*W^T)
__device__ __forceinline__ f16x16 load_b_frag(const _Float16* __restrict__ p) {
  f16x8 b0 = *(const f16x8*)(p);
  f16x8 b1 = *(const f16x8*)(p + 16);
  f16x16 r;
#pragma unroll
  for (int e = 0; e < 8; ++e) { r[e] = b0[e]; r[8 + e] = b1[e]; }
  return r;
}

// ---------------------------------------------------------------------------
// GEMM: C[M,Nreal] = A[M,K](fp32) * W[N,K](f16)^T + bias,  f32 accumulate.
// Block tile 128x128, 256 threads = 8 wave32 waves in 4(M) x 2(N) grid,
// wave tile 32x64 -> 2x4 fragments -> 8 v_wmma_f32_16x16x32_f16 per k-step.
// If batchB > 0, A row m maps to source row (m % batchB)*timeT + (m / batchB)
// (handles x stored [B,T,D] while C is [T,B,H]).
// ---------------------------------------------------------------------------
__global__ __launch_bounds__(256) void gemm_f16w_bias(
    const float* __restrict__ A, const _Float16* __restrict__ Wh,
    const float* __restrict__ bias, float* __restrict__ C,
    int K, int Nreal, int batchB, int timeT) {
  const int lane = threadIdx.x & 31;
  const int wave = threadIdx.x >> 5;
  const int wm = wave >> 1;  // 0..3
  const int wn = wave & 1;   // 0..1
  const int rowBlk = blockIdx.y * 128 + wm * 32;
  const int colBlk = blockIdx.x * 128 + wn * 64;
  const int hs = lane >> 4;  // K-half select
  const int mr = lane & 15;  // row/col within fragment

  size_t arow[2];
#pragma unroll
  for (int i = 0; i < 2; ++i) {
    int m = rowBlk + i * 16 + mr;
    arow[i] = batchB ? ((size_t)(m % batchB) * timeT + (size_t)(m / batchB)) * K
                     : (size_t)m * K;
  }
  const _Float16* wrow[4];
#pragma unroll
  for (int j = 0; j < 4; ++j)
    wrow[j] = Wh + (size_t)(colBlk + j * 16 + mr) * K;

  f32x8 acc[2][4] = {};

  for (int k0 = 0; k0 < K; k0 += 32) {
    const int kk = k0 + hs * 8;
    // speculative prefetch of the next k-tile of A (dropped if OOB)
    __builtin_prefetch(A + arow[0] + kk + 32, 0, 1);
    __builtin_prefetch(A + arow[1] + kk + 32, 0, 1);

    f16x16 af[2];
#pragma unroll
    for (int i = 0; i < 2; ++i) af[i] = load_a_frag(A + arow[i] + kk);
    f16x16 bf[4];
#pragma unroll
    for (int j = 0; j < 4; ++j) bf[j] = load_b_frag(wrow[j] + kk);

#pragma unroll
    for (int i = 0; i < 2; ++i)
#pragma unroll
      for (int j = 0; j < 4; ++j)
        acc[i][j] = __builtin_amdgcn_wmma_f32_16x16x32_f16(
            false, af[i], false, bf[j], (short)0, acc[i][j], false, false);
  }

#pragma unroll
  for (int j = 0; j < 4; ++j) {
    const int n = colBlk + j * 16 + mr;
    if (n >= Nreal) continue;
    const float bv = bias[n];
#pragma unroll
    for (int i = 0; i < 2; ++i) {
#pragma unroll
      for (int r = 0; r < 8; ++r) {
        const int m = rowBlk + i * 16 + hs * 8 + r;
        C[(size_t)m * Nreal + n] = acc[i][j][r] + bv;
      }
    }
  }
}

// ---------------------------------------------------------------------------
// Small-N GEMM for layer 4 (N padded to 16). Block = 256 rows, 8 waves each
// own 32 rows x 16 cols -> 2 fragments, 2 wmma per k-step.
// ---------------------------------------------------------------------------
__global__ __launch_bounds__(256) void gemm_f16w_bias_n16(
    const float* __restrict__ A, const _Float16* __restrict__ Wh,
    const float* __restrict__ bias, float* __restrict__ C,
    int K, int Nreal) {
  const int lane = threadIdx.x & 31;
  const int wave = threadIdx.x >> 5;
  const int rowBlk = blockIdx.x * 256 + wave * 32;
  const int hs = lane >> 4;
  const int mr = lane & 15;

  size_t arow[2];
#pragma unroll
  for (int i = 0; i < 2; ++i)
    arow[i] = (size_t)(rowBlk + i * 16 + mr) * K;
  const _Float16* wrow = Wh + (size_t)mr * K;

  f32x8 acc[2] = {};
  for (int k0 = 0; k0 < K; k0 += 32) {
    const int kk = k0 + hs * 8;
    f16x16 bf = load_b_frag(wrow + kk);
#pragma unroll
    for (int i = 0; i < 2; ++i) {
      f16x16 af = load_a_frag(A + arow[i] + kk);
      acc[i] = __builtin_amdgcn_wmma_f32_16x16x32_f16(
          false, af, false, bf, (short)0, acc[i], false, false);
    }
  }

  const int n = mr;
  if (n < Nreal) {
    const float bv = bias[n];
#pragma unroll
    for (int i = 0; i < 2; ++i)
#pragma unroll
      for (int r = 0; r < 8; ++r) {
        const int m = rowBlk + i * 16 + hs * 8 + r;
        C[(size_t)m * Nreal + n] = acc[i][r] + bv;
      }
  }
}

// ---------------------------------------------------------------------------
// LIF scan over time: one thread per (b, h) neuron, sequential in T.
//   reset = (mem_prev - th > 0);  mem = clip(beta,0,1)*mem_prev + cur - reset*th
//   spk  = (mem - th > 0)
// ---------------------------------------------------------------------------
__global__ void lif_scan(const float* __restrict__ cur,
                         float* __restrict__ spk, float* __restrict__ mem,
                         const float* __restrict__ beta,
                         const float* __restrict__ thr,
                         int layer, int BH, int T) {
  const int j = blockIdx.x * blockDim.x + threadIdx.x;
  if (j >= BH) return;
  float bc = beta[layer];
  bc = fminf(fmaxf(bc, 0.0f), 1.0f);
  const float th = thr[layer];
  float m = 0.0f;
  for (int t = 0; t < T; ++t) {
    const size_t idx = (size_t)t * BH + j;
    const float c = cur[idx];
    const float reset = (m - th > 0.0f) ? 1.0f : 0.0f;
    m = bc * m + c - reset * th;
    spk[idx] = (m - th > 0.0f) ? 1.0f : 0.0f;
    mem[idx] = m;
  }
}

// ---------------------------------------------------------------------------
extern "C" void kernel_launch(void* const* d_in, const int* in_sizes, int n_in,
                              void* d_out, int out_size, void* d_ws,
                              size_t ws_size, hipStream_t stream) {
  (void)in_sizes; (void)n_in; (void)out_size; (void)ws_size;

  const float* x    = (const float*)d_in[0];
  const float* W1   = (const float*)d_in[1];
  const float* b1   = (const float*)d_in[2];
  const float* W2   = (const float*)d_in[3];
  const float* b2   = (const float*)d_in[4];
  const float* W3   = (const float*)d_in[5];
  const float* b3   = (const float*)d_in[6];
  const float* W4   = (const float*)d_in[7];
  const float* b4   = (const float*)d_in[8];
  const float* beta = (const float*)d_in[9];
  const float* thr  = (const float*)d_in[10];

  constexpr int T = 50, B = 256, D = 768, H = 2048, O = 10, NP4 = 16;
  constexpr int M = T * B;  // 12800

  // workspace: f16 weights
  _Float16* W1h = (_Float16*)d_ws;                 // H*D
  _Float16* W2h = W1h + (size_t)H * D;             // H*H
  _Float16* W3h = W2h + (size_t)H * H;             // H*H
  _Float16* W4h = W3h + (size_t)H * H;             // NP4*H (padded)

  // output slices
  float* out = (float*)d_out;
  const size_t S = (size_t)T * B * H, S4 = (size_t)T * B * O;
  float* s1 = out;            float* s2 = out + S;
  float* s3 = out + 2 * S;    float* s4 = out + 3 * S;
  float* m1 = out + 3 * S + S4;  float* m2 = m1 + S;
  float* m3 = m2 + S;            float* m4 = m3 + S;
  float* c1 = m4 + S4;  float* c2 = c1 + S;
  float* c3 = c2 + S;   float* c4 = c3 + S;

  // --- weight conversion -------------------------------------------------
  cvt_f32_to_f16<<<(H * D + 255) / 256, 256, 0, stream>>>(W1, W1h, H * D);
  cvt_f32_to_f16<<<(H * H + 255) / 256, 256, 0, stream>>>(W2, W2h, H * H);
  cvt_f32_to_f16<<<(H * H + 255) / 256, 256, 0, stream>>>(W3, W3h, H * H);
  cvt_pad_f16<<<(NP4 * H + 255) / 256, 256, 0, stream>>>(W4, W4h, H, O, NP4);

  const dim3 gGemm(H / 128, M / 128);  // (16, 100)

  // --- layer 1: x [B,T,D] viewed as [T*B, D] (transposed row map) --------
  gemm_f16w_bias<<<gGemm, 256, 0, stream>>>(x, W1h, b1, c1, D, H, B, T);
  lif_scan<<<(B * H + 255) / 256, 256, 0, stream>>>(c1, s1, m1, beta, thr, 0,
                                                    B * H, T);
  // --- layer 2 -----------------------------------------------------------
  gemm_f16w_bias<<<gGemm, 256, 0, stream>>>(s1, W2h, b2, c2, H, H, 0, 0);
  lif_scan<<<(B * H + 255) / 256, 256, 0, stream>>>(c2, s2, m2, beta, thr, 1,
                                                    B * H, T);
  // --- layer 3 -----------------------------------------------------------
  gemm_f16w_bias<<<gGemm, 256, 0, stream>>>(s2, W3h, b3, c3, H, H, 0, 0);
  lif_scan<<<(B * H + 255) / 256, 256, 0, stream>>>(c3, s3, m3, beta, thr, 2,
                                                    B * H, T);
  // --- layer 4 (N = 10, padded to 16) ------------------------------------
  gemm_f16w_bias_n16<<<M / 256, 256, 0, stream>>>(s3, W4h, b4, c4, H, O);
  lif_scan<<<(B * O + 255) / 256, 256, 0, stream>>>(c4, s4, m4, beta, thr, 3,
                                                    B * O, T);
}
